// CustomLSTM_89910845374597
// MI455X (gfx1250) — matrix operand order repaired and checked by reading, not compiled, not run on hardware
//
#include <hip/hip_runtime.h>
#include <hip/hip_bf16.h>

typedef __attribute__((ext_vector_type(16))) __bf16 v16bf;
typedef __attribute__((ext_vector_type(8)))  __bf16 v8bf;
typedef __attribute__((ext_vector_type(8)))  float  v8f;

#define SEQ   512
#define BATCH 64
#define HDIM  1024
#define GDIM  4096   // 4*HDIM

// ---------------------------------------------------------------------------
// Utility kernels
// ---------------------------------------------------------------------------
__global__ void f32_to_bf16_kernel(const float* __restrict__ in,
                                   __bf16* __restrict__ out, int n) {
  int i = blockIdx.x * blockDim.x + threadIdx.x;
  if (i < n) out[i] = (__bf16)in[i];
}

__global__ void bias_sum_kernel(const float* __restrict__ a,
                                const float* __restrict__ b,
                                float* __restrict__ out, int n) {
  int i = blockIdx.x * blockDim.x + threadIdx.x;
  if (i < n) out[i] = a[i] + b[i];
}

__global__ void init_state_kernel(__bf16* __restrict__ h0,
                                  __bf16* __restrict__ h1,
                                  float* __restrict__ c, int n) {
  int i = blockIdx.x * blockDim.x + threadIdx.x;
  if (i < n) {
    h0[i] = (__bf16)0.0f;
    h1[i] = (__bf16)0.0f;
    c[i]  = 0.0f;
  }
}

// ---------------------------------------------------------------------------
// Fused LSTM step: gates = [x_t | h] @ [W_ih | W_hh]^T + bias, then cell.
// Grid: 64 blocks (one 16-wide hidden strip each), 512 threads (16 waves).
// Wave w -> gate (w>>2), M-tile pair ((w>>1)&1), K-half (w&1):
//   K-half 0: x_t @ W_ih^T  (K=1024)
//   K-half 1: h   @ W_hh^T  (K=1024)
// Partials meet in LDS; the LSTM cell is fused after one barrier.
// Serial WMMA chain per wave: 64; 4 waves/SIMD for latency hiding.
// ---------------------------------------------------------------------------
__device__ __forceinline__ float fast_sigmoid(float x) {
  return 1.0f / (1.0f + __expf(-x));
}
__device__ __forceinline__ float fast_tanh(float x) {
  return 2.0f / (1.0f + __expf(-2.0f * x)) - 1.0f;
}

// Load a 16x32 bf16 A-tile slice for this lane per ISA 7.12.2:
// lanes 0-15 hold K = [0..7] (v0-3) and [16..23] (v4-7) of row M=lane;
// lanes 16-31 hold K = [8..15] and [24..31] of row M=lane-16.
// `base` already points at rowStart + k0 + (lane<16 ? 0 : 8).
__device__ __forceinline__ v16bf load_a_tile(const __bf16* __restrict__ base) {
  v8bf lo = *(const v8bf*)(base);
  v8bf hi = *(const v8bf*)(base + 16);
  v16bf a;
#pragma unroll
  for (int i = 0; i < 8; ++i) { a[i] = lo[i]; a[i + 8] = hi[i]; }
  return a;
}

__global__ __launch_bounds__(512)
void lstm_step_kernel(const __bf16* __restrict__ x_t,     // [64 x 1024] bf16
                      const __bf16* __restrict__ h_in,    // [64 x 1024] bf16
                      const __bf16* __restrict__ Wih,     // [4096 x 1024] bf16 row-major
                      const __bf16* __restrict__ Whh,     // [4096 x 1024] bf16 row-major
                      const float*  __restrict__ bias,    // [4096]  (b_ih+b_hh)
                      float*        __restrict__ c_state, // [64 x 1024] fp32
                      __bf16*       __restrict__ h_out,   // [64 x 1024] bf16 (next step A)
                      float*        __restrict__ hs_f32,  // optional fp32 hidden out
                      __bf16*       __restrict__ hs_bf16, // optional bf16 hidden out
                      float*        __restrict__ hT_out,  // optional final h (fp32)
                      float*        __restrict__ cT_out)  // optional final c (fp32)
{
  // Partial-gate staging: [K-half][gate][batch row][strip col] = 32 KB LDS
  __shared__ float gl[2][4][BATCH][16];

  const int j0   = blockIdx.x * 16;    // hidden-unit strip
  const int tid  = threadIdx.x;
  const int lane = tid & 31;
  const int w    = tid >> 5;           // wave 0..15
  const int kh   = w & 1;              // K-half: 0 = x/W_ih, 1 = h/W_hh
  const int mt0  = ((w >> 1) & 1) * 2; // first of two M-tiles
  const int g    = w >> 2;             // gate 0..3 (i,f,g,o)
  const int lidx = lane & 15;
  const int ao   = (lane < 16) ? 0 : 8;   // A K-offset per lane half
  const int bo   = (lane < 16) ? 0 : 16;  // prefetch K-offset per lane half

  // Select this wave's operand streams (uniform per wave).
  const __bf16* Asrc = (kh == 0) ? x_t : h_in;
  const __bf16* Wsrc = (kh == 0) ? Wih : Whh;

  // B-matrix: column n of the gate strip = row (g*H + j0 + n) of W (K contiguous)
  const __bf16* wrow = Wsrc + (size_t)(g * HDIM + j0 + lidx) * HDIM;
  // A-matrix rows for the two M-tiles
  const __bf16* a0r = Asrc + (size_t)(mt0 * 16 + lidx) * HDIM;
  const __bf16* a1r = Asrc + (size_t)((mt0 + 1) * 16 + lidx) * HDIM;

  v8f acc0 = {};
  v8f acc1 = {};

  // K loop: outer step 128 K-elements, inner fully unrolled (4 WMMA pairs).
  // One branch-free speculative prefetch of the weight stream per chunk
  // (1 KB ahead; address stays inside the workspace, so no bounds check).
  for (int k0 = 0; k0 < HDIM; k0 += 128) {
    __builtin_prefetch(wrow + k0 + 512 + bo, 0, 1);
#pragma unroll
    for (int kk = 0; kk < 128; kk += 32) {
      const int k = k0 + kk;
      v16bf b  = load_a_tile(wrow + k + ao);
      v16bf a0 = load_a_tile(a0r + k + ao);
      v16bf a1 = load_a_tile(a1r + k + ao);
      acc0 = __builtin_amdgcn_wmma_f32_16x16x32_bf16(false, a0, false, b,
                                                     (short)0, acc0, false, false);
      acc1 = __builtin_amdgcn_wmma_f32_16x16x32_bf16(false, a1, false, b,
                                                     (short)0, acc1, false, false);
    }
  }

  // Spill D tiles to LDS. D layout: VGPR v, lane L -> row v + (L>=16 ? 8:0), col L&15.
  const int rbase = (lane & 16) ? 8 : 0;
#pragma unroll
  for (int v = 0; v < 8; ++v) {
    gl[kh][g][mt0 * 16 + rbase + v][lidx]       = acc0[v];
    gl[kh][g][(mt0 + 1) * 16 + rbase + v][lidx] = acc1[v];
  }
  __syncthreads();

  // Fused LSTM cell: 1024 cells (64 rows x 16 strip cols), 2 per thread.
#pragma unroll
  for (int it = 0; it < 2; ++it) {
    int idx = tid + it * 512;
    int m   = idx >> 4;
    int nl  = idx & 15;
    int hid = j0 + nl;
    float iv = gl[0][0][m][nl] + gl[1][0][m][nl] + bias[hid];
    float fv = gl[0][1][m][nl] + gl[1][1][m][nl] + bias[HDIM + hid];
    float gv = gl[0][2][m][nl] + gl[1][2][m][nl] + bias[2 * HDIM + hid];
    float ov = gl[0][3][m][nl] + gl[1][3][m][nl] + bias[3 * HDIM + hid];
    float cp = c_state[m * HDIM + hid];
    float cn = fast_sigmoid(fv) * cp + fast_sigmoid(iv) * fast_tanh(gv);
    float hn = fast_sigmoid(ov) * fast_tanh(cn);
    c_state[m * HDIM + hid] = cn;
    h_out[m * HDIM + hid]   = (__bf16)hn;
    if (hs_f32)  hs_f32[m * HDIM + hid]  = hn;
    if (hs_bf16) hs_bf16[m * HDIM + hid] = (__bf16)hn;
    if (hT_out) {
      hT_out[m * HDIM + hid] = hn;
      cT_out[m * HDIM + hid] = cn;
    }
  }
}

// ---------------------------------------------------------------------------
// Host-side orchestration
// ---------------------------------------------------------------------------
extern "C" void kernel_launch(void* const* d_in, const int* in_sizes, int n_in,
                              void* d_out, int out_size, void* d_ws, size_t ws_size,
                              hipStream_t stream) {
  const float* x     = (const float*)d_in[0];
  const float* W_ih0 = (const float*)d_in[1];
  const float* W_hh0 = (const float*)d_in[2];
  const float* b_ih0 = (const float*)d_in[3];
  const float* b_hh0 = (const float*)d_in[4];
  const float* W_ih1 = (const float*)d_in[5];
  const float* W_hh1 = (const float*)d_in[6];
  const float* b_ih1 = (const float*)d_in[7];
  const float* b_hh1 = (const float*)d_in[8];

  char*  ws  = (char*)d_ws;
  size_t off = 0;
  auto alloc = [&](size_t bytes) -> void* {
    void* p = ws + off;
    off += (bytes + 255) & ~(size_t)255;
    return p;
  };

  const size_t TBH = (size_t)SEQ * BATCH * HDIM;   // 33,554,432
  const size_t BH  = (size_t)BATCH * HDIM;         // 65,536
  const size_t WN  = (size_t)GDIM * HDIM;          // 4,194,304

  __bf16* x_bf    = (__bf16*)alloc(TBH * 2);       // bf16 input
  __bf16* o0_bf   = (__bf16*)alloc(TBH * 2);       // bf16 layer-0 hidden seq
  __bf16* wih0_bf = (__bf16*)alloc(WN * 2);
  __bf16* whh0_bf = (__bf16*)alloc(WN * 2);
  __bf16* wih1_bf = (__bf16*)alloc(WN * 2);
  __bf16* whh1_bf = (__bf16*)alloc(WN * 2);
  float*  bias0   = (float*)alloc(GDIM * 4);
  float*  bias1   = (float*)alloc(GDIM * 4);
  __bf16* hbuf0   = (__bf16*)alloc(BH * 2);        // h double-buffer
  __bf16* hbuf1   = (__bf16*)alloc(BH * 2);
  float*  cbuf    = (float*)alloc(BH * 4);

  // --- precision conversion + bias sums ---
  f32_to_bf16_kernel<<<(int)((WN + 255) / 256), 256, 0, stream>>>(W_ih0, wih0_bf, (int)WN);
  f32_to_bf16_kernel<<<(int)((WN + 255) / 256), 256, 0, stream>>>(W_hh0, whh0_bf, (int)WN);
  f32_to_bf16_kernel<<<(int)((WN + 255) / 256), 256, 0, stream>>>(W_ih1, wih1_bf, (int)WN);
  f32_to_bf16_kernel<<<(int)((WN + 255) / 256), 256, 0, stream>>>(W_hh1, whh1_bf, (int)WN);
  f32_to_bf16_kernel<<<(int)((TBH + 255) / 256), 256, 0, stream>>>(x, x_bf, (int)TBH);
  bias_sum_kernel<<<(GDIM + 255) / 256, 256, 0, stream>>>(b_ih0, b_hh0, bias0, GDIM);
  bias_sum_kernel<<<(GDIM + 255) / 256, 256, 0, stream>>>(b_ih1, b_hh1, bias1, GDIM);

  float* out1 = (float*)d_out;                 // [512, 64, 1024]
  float* h_n  = out1 + TBH;                    // [2, 64, 1024]
  float* c_n  = h_n + 2 * BH;                  // [2, 64, 1024]

  const int stepGrid   = HDIM / 16;            // 64 blocks
  const int stepBlock  = 512;                  // 16 waves

  // --- layer 0 ---
  init_state_kernel<<<(int)((BH + 255) / 256), 256, 0, stream>>>(hbuf0, hbuf1, cbuf, (int)BH);
  for (int t = 0; t < SEQ; ++t) {
    const __bf16* xt = x_bf + (size_t)t * BH;
    __bf16* hi = (t & 1) ? hbuf1 : hbuf0;
    __bf16* ho = (t & 1) ? hbuf0 : hbuf1;
    bool last = (t == SEQ - 1);
    lstm_step_kernel<<<stepGrid, stepBlock, 0, stream>>>(
        xt, hi, wih0_bf, whh0_bf, bias0, cbuf, ho,
        /*hs_f32=*/nullptr,
        /*hs_bf16=*/o0_bf + (size_t)t * BH,
        last ? h_n : nullptr,
        last ? c_n : nullptr);
  }

  // --- layer 1 ---
  init_state_kernel<<<(int)((BH + 255) / 256), 256, 0, stream>>>(hbuf0, hbuf1, cbuf, (int)BH);
  for (int t = 0; t < SEQ; ++t) {
    const __bf16* xt = o0_bf + (size_t)t * BH;
    __bf16* hi = (t & 1) ? hbuf1 : hbuf0;
    __bf16* ho = (t & 1) ? hbuf0 : hbuf1;
    bool last = (t == SEQ - 1);
    lstm_step_kernel<<<stepGrid, stepBlock, 0, stream>>>(
        xt, hi, wih1_bf, whh1_bf, bias1, cbuf, ho,
        /*hs_f32=*/out1 + (size_t)t * BH,
        /*hs_bf16=*/nullptr,
        last ? h_n + BH : nullptr,
        last ? c_n + BH : nullptr);
  }
}
